// OCRLayer_34376918238051
// MI455X (gfx1250) — matrix operand order repaired and checked by reading, not compiled, not run on hardware
//
#include <hip/hip_runtime.h>
#include <hip/hip_bf16.h>
#include <stdint.h>

// Problem constants (from reference)
#define NN 100000
#define CC 256
#define HH 8
#define KK 20
#define BB 4
#define DD 32
#define INV_SQRT_D 0.17677669529663687f

typedef __attribute__((ext_vector_type(16))) __bf16 v16bf;
typedef __attribute__((ext_vector_type(8)))  float  v8f;
typedef __attribute__((ext_vector_type(4)))  unsigned int u32x4;

union FragU { u32x4 u[2]; v16bf v; };
union Pack8 { __bf16 h[8]; u32x4 u; };

// LDS tile: 16 rows x 256 bf16, padded to 264 (528B row stride -> row r maps to
// bank 4r, so 16 lanes doing ds_load_b128 cover all 64 banks conflict-free)
#define LDA 264
#define TILE_ELEMS (16 * LDA)
// Per wave: 3 buffers (feat, q/ao aliased, out) x 2 row-tiles (M=32)
#define WAVE_TILES 6
#define WAVE_ELEMS (WAVE_TILES * TILE_ELEMS)

// ---------------------------------------------------------------------------
// WMMA fragment loaders (per CDNA5 ISA 16-bit A/B layouts, wave32)
// A: lanes 0-15 hold M=lane rows; per-lane 16 bf16 at K = kbase + 8*half + {0..7,16..23}
// B: lane holds column N=lane%16; 16 bf16 at K = kbase + 8*half + {0..7,16..23}
// ---------------------------------------------------------------------------
__device__ __forceinline__ v16bf load_a_frag(const __bf16* tile, int m, int kbase, int half) {
  const __bf16* p = tile + m * LDA + kbase + half * 8;
  FragU f;
  f.u[0] = *(const u32x4*)(p);
  f.u[1] = *(const u32x4*)(p + 16);
  return f.v;
}

__device__ __forceinline__ v16bf load_b_frag(const __bf16* W, int ldw, int row, int kbase, int half) {
  const __bf16* p = W + (size_t)row * ldw + kbase + half * 8;
  FragU f;
  f.u[0] = *(const u32x4*)(p);
  f.u[1] = *(const u32x4*)(p + 16);
  return f.v;
}

__device__ __forceinline__ v8f wmma_bf16(v16bf a, v16bf b, v8f c) {
  return __builtin_amdgcn_wmma_f32_16x16x32_bf16(false, a, false, b, (short)0, c, false, false);
}

// ---------------------------------------------------------------------------
// Kernel 0: convert weights fp32 -> bf16 (wq pre-scaled by 1/sqrt(D)), zero ctx
// ---------------------------------------------------------------------------
__global__ __launch_bounds__(256) void prep_kernel(
    const float* __restrict__ in_proj_w, const float* __restrict__ out_proj_w,
    const float* __restrict__ bottleneck_w,
    __bf16* __restrict__ wq_bf, __bf16* __restrict__ wo_bf, __bf16* __restrict__ wb_bf,
    float* __restrict__ ctx) {
  int idx = blockIdx.x * 256 + threadIdx.x;
  if (idx < 65536) {
    wq_bf[idx] = (__bf16)(in_proj_w[idx] * INV_SQRT_D);
  } else if (idx < 131072) {
    int i = idx - 65536;
    wo_bf[i] = (__bf16)(out_proj_w[i]);
  } else if (idx < 262144) {
    int i = idx - 131072;
    wb_bf[i] = (__bf16)(bottleneck_w[i]);
  } else if (idx < 262144 + BB * KK * CC) {
    ctx[idx - 262144] = 0.0f;
  }
}

// ---------------------------------------------------------------------------
// Kernel 1: softmax(probs) + segmented reduction into context[b][k][c]
// Block = 256 threads, chunk = 512 rows. Thread t owns column c=t and keeps
// 20 register accumulators; sorted batch_indices -> flush on batch change
// via global float atomics.
// ---------------------------------------------------------------------------
__global__ __launch_bounds__(256) void softmax_ctx_kernel(
    const float* __restrict__ probs, const float* __restrict__ features,
    const int* __restrict__ bidx, float* __restrict__ ctx) {
  __shared__ float p_lds[512][KK];
  int t = threadIdx.x;
  int base = blockIdx.x * 512;

  for (int rr = 0; rr < 2; ++rr) {
    int rl = t + rr * 256;
    int row = base + rl;
    if (row < NN) {
      float v[KK];
      float m = -1e30f;
#pragma unroll
      for (int k = 0; k < KK; ++k) { v[k] = probs[(size_t)row * KK + k]; m = fmaxf(m, v[k]); }
      float s = 0.f;
#pragma unroll
      for (int k = 0; k < KK; ++k) { v[k] = __expf(v[k] - m); s += v[k]; }
      float inv = 1.f / s;
#pragma unroll
      for (int k = 0; k < KK; ++k) p_lds[rl][k] = v[k] * inv;
    } else {
#pragma unroll
      for (int k = 0; k < KK; ++k) p_lds[rl][k] = 0.f;
    }
  }
  __syncthreads();

  float acc[KK];
#pragma unroll
  for (int k = 0; k < KK; ++k) acc[k] = 0.f;
  int cur_b = -1;
  for (int rl = 0; rl < 512; ++rl) {
    int row = base + rl;
    if (row >= NN) break;
    int b = bidx[row];
    if (b != cur_b) {
      if (cur_b >= 0) {
#pragma unroll
        for (int k = 0; k < KK; ++k)
          atomicAdd(&ctx[(cur_b * KK + k) * CC + t], acc[k]);
      }
      cur_b = b;
#pragma unroll
      for (int k = 0; k < KK; ++k) acc[k] = 0.f;
    }
    float f = features[(size_t)row * CC + t];
#pragma unroll
    for (int k = 0; k < KK; ++k) acc[k] += p_lds[rl][k] * f;
  }
  if (cur_b >= 0) {
#pragma unroll
    for (int k = 0; k < KK; ++k)
      atomicAdd(&ctx[(cur_b * KK + k) * CC + t], acc[k]);
  }
}

// ---------------------------------------------------------------------------
// Kernel 2: kproj/vproj = context @ {wk,wv}^T + bias  (80 x 256 each; tiny)
// ---------------------------------------------------------------------------
__global__ __launch_bounds__(256) void kvproj_kernel(
    const float* __restrict__ ctx, const float* __restrict__ in_proj_w,
    const float* __restrict__ in_proj_b,
    float* __restrict__ kproj, float* __restrict__ vproj) {
  __shared__ float crow[CC];
  int row = blockIdx.x;      // 0..79  (b*K + k)
  int j = threadIdx.x;       // 0..255
  crow[j] = ctx[row * CC + j];
  __syncthreads();
  const float* wk = in_proj_w + (size_t)(CC + j) * CC;
  const float* wv = in_proj_w + (size_t)(2 * CC + j) * CC;
  float ka = 0.f, va = 0.f;
#pragma unroll 4
  for (int c = 0; c < CC; ++c) {
    float cv = crow[c];
    ka += cv * wk[c];
    va += cv * wv[c];
  }
  kproj[row * CC + j] = ka + in_proj_b[CC + j];
  vproj[row * CC + j] = va + in_proj_b[2 * CC + j];
}

// ---------------------------------------------------------------------------
// Kernel 3: fused Q-proj -> attention -> out-proj -> bottleneck
// 128 threads = 4 waves; each wave owns a 32-row (2 x 16) A tile so every
// B fragment fetched from L2 feeds two WMMAs (halves weight traffic).
// LDS buffers per wave: feat[2 tiles] | q/ao[2 tiles, aliased] | out[2 tiles].
// ---------------------------------------------------------------------------
__global__ __launch_bounds__(128) void fused_main_kernel(
    const float* __restrict__ features, const int* __restrict__ bidx,
    const float* __restrict__ in_proj_b, const float* __restrict__ out_proj_b,
    const __bf16* __restrict__ wq_bf, const __bf16* __restrict__ wo_bf,
    const __bf16* __restrict__ wb_bf,
    const float* __restrict__ kproj, const float* __restrict__ vproj,
    float* __restrict__ out) {
  extern __shared__ char dynsmem[];
  __bf16* smem = (__bf16*)dynsmem;

  int tid  = threadIdx.x;
  int wave = tid >> 5;
  int lane = tid & 31;
  int half = lane >> 4;
  int l16  = lane & 15;

  __bf16* featb = smem + wave * WAVE_ELEMS;          // 2 tiles
  __bf16* qab   = featb + 2 * TILE_ELEMS;            // 2 tiles: q, then attn_out
  __bf16* outb  = featb + 4 * TILE_ELEMS;            // 2 tiles: attention output

  int rowbase = blockIdx.x * 128 + wave * 32;

  // Warm L2->L0 for this tile's K/V rows (global_prefetch_b8); 20KB each.
  {
    int gr0 = rowbase; if (gr0 > NN - 1) gr0 = NN - 1;
    int b0 = bidx[gr0];
    const char* kp = (const char*)(kproj + b0 * KK * CC);
    const char* vp = (const char*)(vproj + b0 * KK * CC);
#pragma unroll
    for (int q = 0; q < 5; ++q) {
      __builtin_prefetch(kp + (q * 32 + lane) * 128, 0, 3);
      __builtin_prefetch(vp + (q * 32 + lane) * 128, 0, 3);
    }
  }

  // ---- Phase 1: 32 feature rows -> bf16 LDS (ds_store_b128 per row/lane)
  for (int r = 0; r < 32; ++r) {
    int gr = rowbase + r; if (gr > NN - 1) gr = NN - 1;
    const float4* src = (const float4*)(features + (size_t)gr * CC) + lane * 2;
    float4 f0 = src[0];
    float4 f1 = src[1];
    Pack8 pk;
    pk.h[0] = (__bf16)f0.x; pk.h[1] = (__bf16)f0.y;
    pk.h[2] = (__bf16)f0.z; pk.h[3] = (__bf16)f0.w;
    pk.h[4] = (__bf16)f1.x; pk.h[5] = (__bf16)f1.y;
    pk.h[6] = (__bf16)f1.z; pk.h[7] = (__bf16)f1.w;
    *(u32x4*)(featb + (r >> 4) * TILE_ELEMS + (r & 15) * LDA + lane * 8) = pk.u;
  }
  __syncthreads();

  // ---- Phase 2: Q = feat @ (wq*scale)^T + bq*scale   (bf16 WMMA, f32 acc)
  for (int j = 0; j < 16; ++j) {
    v8f acc0 = {0.f, 0.f, 0.f, 0.f, 0.f, 0.f, 0.f, 0.f};
    v8f acc1 = acc0;
#pragma unroll
    for (int kt = 0; kt < 8; ++kt) {
      v16bf bfr = load_b_frag(wq_bf, CC, j * 16 + l16, kt * 32, half);
      v16bf a0 = load_a_frag(featb, l16, kt * 32, half);
      v16bf a1 = load_a_frag(featb + TILE_ELEMS, l16, kt * 32, half);
      acc0 = wmma_bf16(a0, bfr, acc0);
      acc1 = wmma_bf16(a1, bfr, acc1);
    }
    float bias = in_proj_b[j * 16 + l16] * INV_SQRT_D;
#pragma unroll
    for (int i = 0; i < 8; ++i) {
      qab[(i + 8 * half) * LDA + j * 16 + l16] = (__bf16)(acc0[i] + bias);
      qab[TILE_ELEMS + (i + 8 * half) * LDA + j * 16 + l16] = (__bf16)(acc1[i] + bias);
    }
  }
  __syncthreads();

  // ---- Phase 3: attention (K=20 keys, D=32 per head) — VALU
  // 32 rows x 8 heads = 256 (row,head) pairs per wave, 8 per lane.
  for (int pp = 0; pp < 8; ++pp) {
    int pairIdx = pp * 32 + lane;
    int r = pairIdx >> 3;                // 0..31
    int h = pairIdx & 7;
    int gr = rowbase + r; if (gr > NN - 1) gr = NN - 1;
    int b = bidx[gr];
    const __bf16* qrow = qab + (r >> 4) * TILE_ELEMS + (r & 15) * LDA + h * DD;

    float qv[DD];
#pragma unroll
    for (int d = 0; d < DD; ++d) qv[d] = (float)qrow[d];

    float sc[KK];
    float m = -1e30f;
    for (int k = 0; k < KK; ++k) {
      const float* kp = kproj + ((b * KK + k) * CC + h * DD);
      float s = 0.f;
#pragma unroll
      for (int d = 0; d < DD; ++d) s += qv[d] * kp[d];
      sc[k] = s;
      m = fmaxf(m, s);
    }
    float sum = 0.f;
#pragma unroll
    for (int k = 0; k < KK; ++k) { sc[k] = __expf(sc[k] - m); sum += sc[k]; }
    float inv = 1.f / sum;

    float o[DD];
#pragma unroll
    for (int d = 0; d < DD; ++d) o[d] = 0.f;
    for (int k = 0; k < KK; ++k) {
      const float* vp = vproj + ((b * KK + k) * CC + h * DD);
      float a = sc[k] * inv;
#pragma unroll
      for (int d = 0; d < DD; ++d) o[d] += a * vp[d];
    }
    __bf16* orow = outb + (r >> 4) * TILE_ELEMS + (r & 15) * LDA + h * DD;
#pragma unroll
    for (int d = 0; d < DD; ++d) orow[d] = (__bf16)o[d];
  }
  __syncthreads();

  // ---- Phase 4: attn_out = out @ wo^T + bo   (overwrites q buffer)
  for (int j = 0; j < 16; ++j) {
    v8f acc0 = {0.f, 0.f, 0.f, 0.f, 0.f, 0.f, 0.f, 0.f};
    v8f acc1 = acc0;
#pragma unroll
    for (int kt = 0; kt < 8; ++kt) {
      v16bf bfr = load_b_frag(wo_bf, CC, j * 16 + l16, kt * 32, half);
      v16bf a0 = load_a_frag(outb, l16, kt * 32, half);
      v16bf a1 = load_a_frag(outb + TILE_ELEMS, l16, kt * 32, half);
      acc0 = wmma_bf16(a0, bfr, acc0);
      acc1 = wmma_bf16(a1, bfr, acc1);
    }
    float bias = out_proj_b[j * 16 + l16];
#pragma unroll
    for (int i = 0; i < 8; ++i) {
      qab[(i + 8 * half) * LDA + j * 16 + l16] = (__bf16)(acc0[i] + bias);
      qab[TILE_ELEMS + (i + 8 * half) * LDA + j * 16 + l16] = (__bf16)(acc1[i] + bias);
    }
  }
  __syncthreads();

  // ---- Phase 5: final = [feat, attn_out] @ wb^T   (K = 512)
  for (int j = 0; j < 16; ++j) {
    v8f acc0 = {0.f, 0.f, 0.f, 0.f, 0.f, 0.f, 0.f, 0.f};
    v8f acc1 = acc0;
#pragma unroll
    for (int kt = 0; kt < 8; ++kt) {
      v16bf bfr = load_b_frag(wb_bf, 2 * CC, j * 16 + l16, kt * 32, half);
      v16bf a0 = load_a_frag(featb, l16, kt * 32, half);
      v16bf a1 = load_a_frag(featb + TILE_ELEMS, l16, kt * 32, half);
      acc0 = wmma_bf16(a0, bfr, acc0);
      acc1 = wmma_bf16(a1, bfr, acc1);
    }
#pragma unroll
    for (int kt = 0; kt < 8; ++kt) {
      v16bf bfr = load_b_frag(wb_bf, 2 * CC, j * 16 + l16, CC + kt * 32, half);
      v16bf a0 = load_a_frag(qab, l16, kt * 32, half);
      v16bf a1 = load_a_frag(qab + TILE_ELEMS, l16, kt * 32, half);
      acc0 = wmma_bf16(a0, bfr, acc0);
      acc1 = wmma_bf16(a1, bfr, acc1);
    }
#pragma unroll
    for (int i = 0; i < 8; ++i) {
      int gr0 = rowbase + i + 8 * half;
      int gr1 = rowbase + 16 + i + 8 * half;
      if (gr0 < NN) out[(size_t)gr0 * CC + j * 16 + l16] = acc0[i];
      if (gr1 < NN) out[(size_t)gr1 * CC + j * 16 + l16] = acc1[i];
    }
  }
}

// ---------------------------------------------------------------------------
// Host launcher
// ---------------------------------------------------------------------------
extern "C" void kernel_launch(void* const* d_in, const int* in_sizes, int n_in,
                              void* d_out, int out_size, void* d_ws, size_t ws_size,
                              hipStream_t stream) {
  const float* features     = (const float*)d_in[0];
  const float* probs        = (const float*)d_in[1];
  const float* in_proj_w    = (const float*)d_in[2];
  const float* in_proj_b    = (const float*)d_in[3];
  const float* out_proj_w   = (const float*)d_in[4];
  const float* out_proj_b   = (const float*)d_in[5];
  const float* bottleneck_w = (const float*)d_in[6];
  const int*   batch_idx    = (const int*)d_in[7];
  float* out = (float*)d_out;

  char* ws = (char*)d_ws;
  // Workspace layout (bytes)
  float*  ctx   = (float*) (ws + 0);                         //  80*256*4  = 81920
  float*  kproj = (float*) (ws + 81920);                     //  81920
  float*  vproj = (float*) (ws + 163840);                    //  81920
  __bf16* wq_bf = (__bf16*)(ws + 245760);                    //  65536*2   = 131072
  __bf16* wo_bf = (__bf16*)(ws + 376832);                    //  131072
  __bf16* wb_bf = (__bf16*)(ws + 507904);                    //  131072*2  = 262144
  // total: 770048 bytes

  // 0) weight conversion + context zeroing
  prep_kernel<<<dim3((262144 + BB * KK * CC + 255) / 256), dim3(256), 0, stream>>>(
      in_proj_w, out_proj_w, bottleneck_w, wq_bf, wo_bf, wb_bf, ctx);

  // 1) softmax + segmented context reduction
  softmax_ctx_kernel<<<dim3((NN + 511) / 512), dim3(256), 0, stream>>>(
      probs, features, batch_idx, ctx);

  // 2) k/v projections (80 rows)
  kvproj_kernel<<<dim3(BB * KK), dim3(256), 0, stream>>>(
      ctx, in_proj_w, in_proj_b, kproj, vproj);

  // 3) fused main pipeline: 128 rows/block, 202752B dynamic LDS
  size_t lds_bytes = (size_t)4 * WAVE_ELEMS * sizeof(__bf16);
  fused_main_kernel<<<dim3((NN + 127) / 128), dim3(128), lds_bytes, stream>>>(
      features, batch_idx, in_proj_b, out_proj_b,
      wq_bf, wo_bf, wb_bf, kproj, vproj, out);
}
